// SelectiveContextAttention_35656818492108
// MI455X (gfx1250) — matrix operand order repaired and checked by reading, not compiled, not run on hardware
//
#include <hip/hip_runtime.h>
#include <hip/hip_bf16.h>
#include <math.h>
#include <stdint.h>

// ---------------------------------------------------------------------------
// SelectiveContextAttention forward, CDNA5 (gfx1250), wave32.
// bf16x3 split-precision WMMA (v_wmma_f32_16x16x32_bf16) for all GEMMs and
// the fused two-softmax flash attention. Tiles are staged global->LDS with
// GLOBAL_LOAD_ASYNC_TO_LDS_B128 (ASYNCcnt DMA), double-buffered so the DMA
// overlaps the WMMA stream. All fragment reads are contiguous ds_load_b128.
// ---------------------------------------------------------------------------

typedef __attribute__((ext_vector_type(16))) __bf16 v16bf;
typedef __attribute__((ext_vector_type(8)))  __bf16 v8bf;
typedef __attribute__((ext_vector_type(8)))  float  v8f;

#define WMMA_BF16(a, b, c) \
  __builtin_amdgcn_wmma_f32_16x16x32_bf16(false, (a), false, (b), (short)0, (c), false, false)

static constexpr int  Bn   = 2;
static constexpr int  Tn   = 2048;
static constexpr int  Hc   = 768;
static constexpr int  H3   = 2304;
static constexpr int  NHn  = 12;
static constexpr int  HDd  = 64;
static constexpr int  LWn  = 256;
static constexpr int  NTOK = Bn * Tn;          // 4096
static constexpr float NEGF = -1e9f;

// Async DMA: 16 bytes global -> LDS, tracked by ASYNCcnt.
__device__ __forceinline__ void async_cp16(void* lds, const void* gm) {
  asm volatile("global_load_async_to_lds_b128 %0, %1, off"
               :: "v"((unsigned)(uintptr_t)lds), "v"(gm) : "memory");
}
__device__ __forceinline__ void wait_async0() {
  asm volatile("s_wait_asynccnt 0x0" ::: "memory");
}

// two contiguous 16B chunks -> one 16-element bf16 fragment
__device__ __forceinline__ v16bf ld_frag2(const __bf16* p0, const __bf16* p1) {
  v8bf a = *(const v8bf*)p0;
  v8bf b = *(const v8bf*)p1;
  return __builtin_shufflevector(a, b, 0, 1, 2, 3, 4, 5, 6, 7,
                                       8, 9, 10, 11, 12, 13, 14, 15);
}

// ---------------------------------------------------------------------------
// Split fp32 -> (hi, lo) bf16, row-major
// ---------------------------------------------------------------------------
__global__ void k_split(const float* __restrict__ src, __bf16* __restrict__ hi,
                        __bf16* __restrict__ lo, int n) {
  int i = blockIdx.x * 256 + threadIdx.x;
  if (i < n) {
    float x = src[i];
    __bf16 h = (__bf16)x;
    hi[i] = h;
    lo[i] = (__bf16)(x - (float)h);
  }
}

// Split fp32 [K][N] -> (hi, lo) bf16 TRANSPOSED [N][K]
__global__ void k_split_t(const float* __restrict__ src, __bf16* __restrict__ hiT,
                          __bf16* __restrict__ loT, int K, int N) {
  int idx = blockIdx.x * 256 + threadIdx.x;
  if (idx >= K * N) return;
  int k = idx / N, n = idx % N;
  float x = src[idx];
  __bf16 h = (__bf16)x;
  hiT[(size_t)n * K + k] = h;
  loT[(size_t)n * K + k] = (__bf16)(x - (float)h);
}

// ---------------------------------------------------------------------------
// Generic bf16x3 GEMM: C[M,N] = A[M,K] @ W[K,N] + bias, W given TRANSPOSED.
// Block tile 128x64, BK=32, 256 threads (8 waves), wave tile 32x32.
// Double-buffered async staging; fragments read as ds_load_b128.
// ---------------------------------------------------------------------------
__launch_bounds__(256)
__global__ void k_gemm(const __bf16* __restrict__ Ah, const __bf16* __restrict__ Al,
                       const __bf16* __restrict__ WtH, const __bf16* __restrict__ WtL, // [N][K]
                       const float* __restrict__ bias, float* __restrict__ C,
                       int M, int N, int K) {
  alignas(16) __shared__ __bf16 sAh[2][128][40];
  alignas(16) __shared__ __bf16 sAl[2][128][40];
  alignas(16) __shared__ __bf16 sWh[2][64][40];   // [n][k]
  alignas(16) __shared__ __bf16 sWl[2][64][40];

  const int tid  = threadIdx.x;
  const int wid  = tid >> 5;
  const int lane = tid & 31;
  const int lm   = lane & 15;
  const int hv   = lane >> 4;
  const int m0   = blockIdx.y * 128;
  const int n0   = blockIdx.x * 64;
  const int wr   = (wid >> 1) * 32;     // wave row within block tile
  const int wc   = (wid & 1) * 32;      // wave col within block tile

  auto issue = [&](int k0, int pb) {
    // A tile 128x32: 512 16B chunks, 2 per thread (x hi/lo)
    for (int t = tid; t < 512; t += 256) {
      int r = t >> 2, c = (t & 3) * 8;
      size_t g = (size_t)(m0 + r) * K + (k0 + c);
      async_cp16(&sAh[pb][r][c], &Ah[g]);
      async_cp16(&sAl[pb][r][c], &Al[g]);
    }
    // Wt tile 64x32: 256 16B chunks, 1 per thread (x hi/lo)
    {
      int n = tid >> 2, c = (tid & 3) * 8;
      size_t g = (size_t)(n0 + n) * K + (k0 + c);
      async_cp16(&sWh[pb][n][c], &WtH[g]);
      async_cp16(&sWl[pb][n][c], &WtL[g]);
    }
  };

  v8f acc[2][2] = {};
  issue(0, 0);
  int p = 0;

  for (int k0 = 0; k0 < K; k0 += 32) {
    wait_async0();
    __syncthreads();
    if (k0 + 32 < K) issue(k0 + 32, p ^ 1);

    v16bf ah[2], al[2];
#pragma unroll
    for (int mt = 0; mt < 2; ++mt) {
      int rb = wr + mt * 16 + lm;
      ah[mt] = ld_frag2(&sAh[p][rb][hv * 8], &sAh[p][rb][16 + hv * 8]);
      al[mt] = ld_frag2(&sAl[p][rb][hv * 8], &sAl[p][rb][16 + hv * 8]);
    }
    v16bf bh[2], bl[2];
#pragma unroll
    for (int nt = 0; nt < 2; ++nt) {
      int cb = wc + nt * 16 + lm;
      bh[nt] = ld_frag2(&sWh[p][cb][hv * 8], &sWh[p][cb][16 + hv * 8]);
      bl[nt] = ld_frag2(&sWl[p][cb][hv * 8], &sWl[p][cb][16 + hv * 8]);
    }
#pragma unroll
    for (int mt = 0; mt < 2; ++mt)
#pragma unroll
      for (int nt = 0; nt < 2; ++nt) {
        acc[mt][nt] = WMMA_BF16(ah[mt], bh[nt], acc[mt][nt]);
        acc[mt][nt] = WMMA_BF16(ah[mt], bl[nt], acc[mt][nt]);
        acc[mt][nt] = WMMA_BF16(al[mt], bh[nt], acc[mt][nt]);
      }
    p ^= 1;
  }

  // Epilogue: D layout VGPR v -> row v (lanes 0-15) / v+8 (lanes 16-31)
#pragma unroll
  for (int mt = 0; mt < 2; ++mt)
#pragma unroll
    for (int nt = 0; nt < 2; ++nt)
#pragma unroll
      for (int v = 0; v < 8; ++v) {
        int row = m0 + wr + mt * 16 + v + 8 * hv;
        int col = n0 + wc + nt * 16 + lm;
        float r = acc[mt][nt][v] + (bias ? bias[col] : 0.f);
        C[(size_t)row * N + col] = r;
      }
}

// ---------------------------------------------------------------------------
// Repack qkv fp32 [NTOK, 3H] -> per-head hi/lo bf16.
// Q, K row-major [B,NH,T,HD]; V TRANSPOSED per head [B,NH,HD,T].
// ---------------------------------------------------------------------------
__global__ void k_repack_qkv(const float* __restrict__ qkv,
                             __bf16* qh, __bf16* ql, __bf16* kh, __bf16* kl,
                             __bf16* vth, __bf16* vtl) {
  size_t idx = (size_t)blockIdx.x * 256 + threadIdx.x;
  size_t total = (size_t)NTOK * H3;
  if (idx >= total) return;
  int    col = (int)(idx % H3);
  size_t bt  = idx / H3;
  int b = (int)(bt / Tn), t = (int)(bt % Tn);
  int which = col / Hc;
  int cc    = col % Hc;
  int h = cc / HDd, d = cc % HDd;
  float  x  = qkv[idx];
  __bf16 hi = (__bf16)x;
  __bf16 lo = (__bf16)(x - (float)hi);
  size_t bh = (size_t)b * NHn + h;
  if (which == 2) {
    size_t dst = (bh * HDd + d) * Tn + t;      // [d][t]
    vth[dst] = hi; vtl[dst] = lo;
  } else {
    size_t dst = (bh * Tn + t) * HDd + d;      // [t][d]
    if (which == 0) { qh[dst] = hi; ql[dst] = lo; }
    else            { kh[dst] = hi; kl[dst] = lo; }
  }
}

// ---------------------------------------------------------------------------
// sel = sigmoid(gelu(sel1) @ w_sel2 + b_sel2), exact erf gelu. [NTOK]
// ---------------------------------------------------------------------------
__global__ void k_sel2(const float* __restrict__ sel1, const float* __restrict__ w2,
                       const float* __restrict__ b2, float* __restrict__ sel, int M) {
  int row = blockIdx.x * 256 + threadIdx.x;
  if (row >= M) return;
  float acc = b2[0];
  for (int k = 0; k < Hc / 4; ++k) {
    float x = sel1[(size_t)row * (Hc / 4) + k];
    float g = 0.5f * x * (1.0f + erff(x * 0.70710678118654752f));
    acc += g * w2[k];
  }
  sel[row] = 1.0f / (1.0f + __expf(-acc));
}

// ---------------------------------------------------------------------------
// Fused attention: per (b,h, 64-query tile). 4 waves, each owns 16 q-rows.
// Two online softmaxes (local window, selectivity-modulated), P.V via WMMA.
// K staged row-major [key][dim] (pitch 72); V staged [dim][key] (pitch 40)
// via async DMA from the pre-transposed global V. Double-buffered.
// ---------------------------------------------------------------------------
__launch_bounds__(128)
__global__ void k_attn(const __bf16* __restrict__ qh, const __bf16* __restrict__ ql,
                       const __bf16* __restrict__ kh, const __bf16* __restrict__ kl,
                       const __bf16* __restrict__ vth, const __bf16* __restrict__ vtl,
                       const float* __restrict__ sel,
                       const float* __restrict__ lwp, const float* __restrict__ gwp,
                       __bf16* __restrict__ oh, __bf16* __restrict__ ol) {
  alignas(16) __shared__ __bf16 sKh[2][32][72], sKl[2][32][72];   // [key][dim]
  alignas(16) __shared__ __bf16 sVh[2][64][40], sVl[2][64][40];   // [dim][key]
  alignas(16) __shared__ __bf16 sPh[4][16][40], sPl[4][16][40];

  const int tid  = threadIdx.x;
  const int wid  = tid >> 5;
  const int lane = tid & 31;
  const int lm   = lane & 15;
  const int hv   = lane >> 4;
  const int q0   = blockIdx.x * 64;
  const int bhid = blockIdx.y;            // b*NH + h
  const int b    = bhid / NHn;
  const int h    = bhid % NHn;
  const size_t headoff = (size_t)bhid * Tn * HDd;
  const size_t vtoff   = (size_t)bhid * HDd * Tn;
  const float scale = 0.125f;             // 64^-0.5

  auto issue = [&](int j0, int pb) {
    // K tile 32x64: 256 16B chunks, 2 per thread (x hi/lo)
    for (int t = tid; t < 256; t += 128) {
      int r = t >> 3, c = (t & 7) * 8;
      size_t g = headoff + (size_t)(j0 + r) * HDd + c;
      async_cp16(&sKh[pb][r][c], &kh[g]);
      async_cp16(&sKl[pb][r][c], &kl[g]);
    }
    // Vt tile 64x32: 256 16B chunks, 2 per thread (x hi/lo)
    for (int t = tid; t < 256; t += 128) {
      int d = t >> 2, c = (t & 3) * 8;
      size_t g = vtoff + (size_t)d * Tn + (j0 + c);
      async_cp16(&sVh[pb][d][c], &vth[g]);
      async_cp16(&sVl[pb][d][c], &vtl[g]);
    }
  };

  // Q fragments (persistent): rows q0 + wid*16 + m, dims 0..63 (2 k-chunks)
  v16bf fqh[2], fql[2];
  {
    size_t base = headoff + (size_t)(q0 + wid * 16 + lm) * HDd;
#pragma unroll
    for (int c = 0; c < 2; ++c) {
      fqh[c] = ld_frag2(&qh[base + c * 32 + hv * 8], &qh[base + c * 32 + 16 + hv * 8]);
      fql[c] = ld_frag2(&ql[base + c * 32 + hv * 8], &ql[base + c * 32 + 16 + hv * 8]);
    }
  }

  v8f accL[4] = {}, accM[4] = {};
  float mL[8], lL[8], mM[8], lM[8];
#pragma unroll
  for (int v = 0; v < 8; ++v) { mL[v] = -3.0e38f; lL[v] = 0.f; mM[v] = -3.0e38f; lM[v] = 0.f; }

  const int iRowBase = q0 + wid * 16;
  const int jend = q0 + 64;               // causal bound (exclusive)

  issue(0, 0);
  int p = 0;

  for (int j0 = 0; j0 < jend; j0 += 32) {
    wait_async0();
    __syncthreads();
    if (j0 + 32 < jend) issue(j0 + 32, p ^ 1);

    // S = Q @ K^T for two 16-key subtiles (bf16x3, K=64 over head dim)
    v8f s0 = {}, s1 = {};
#pragma unroll
    for (int st = 0; st < 2; ++st) {
      v8f sa = {};
      int kr = st * 16 + lm;               // key (column of S)
#pragma unroll
      for (int c = 0; c < 2; ++c) {
        v16bf bhf = ld_frag2(&sKh[p][kr][c * 32 + hv * 8], &sKh[p][kr][c * 32 + 16 + hv * 8]);
        v16bf blf = ld_frag2(&sKl[p][kr][c * 32 + hv * 8], &sKl[p][kr][c * 32 + 16 + hv * 8]);
        sa = WMMA_BF16(fqh[c], bhf, sa);
        sa = WMMA_BF16(fqh[c], blf, sa);
        sa = WMMA_BF16(fql[c], bhf, sa);
      }
      if (st == 0) s0 = sa; else s1 = sa;
    }

    // V fragments as B operand (32 keys x 64 dims -> 4 dim-tiles), hi/lo
    v16bf fvh[4], fvl[4];
#pragma unroll
    for (int nt = 0; nt < 4; ++nt) {
      int cb = nt * 16 + lm;         // dim
      fvh[nt] = ld_frag2(&sVh[p][cb][hv * 8], &sVh[p][cb][16 + hv * 8]);
      fvl[nt] = ld_frag2(&sVl[p][cb][hv * 8], &sVl[p][cb][16 + hv * 8]);
    }

    const float g0 = 0.2f + 0.8f * sel[(size_t)b * Tn + j0 + lm];
    const float g1 = 0.2f + 0.8f * sel[(size_t)b * Tn + j0 + 16 + lm];

    auto flavor = [&](v8f (&accS)[4], float (&mS)[8], float (&lS)[8], bool isLocal) {
      float p0v[8], p1v[8], tmax[8];
#pragma unroll
      for (int v = 0; v < 8; ++v) {
        int i  = iRowBase + v + 8 * hv;
        int ja = j0 + lm;
        int jb = j0 + 16 + lm;
        float r0 = s0[v] * scale;
        float r1 = s1[v] * scale;
        float a, c;
        if (isLocal) {
          a = (ja > i || ja < i - LWn) ? NEGF : r0;
          c = (jb > i || jb < i - LWn) ? NEGF : r1;
        } else {
          a = ((ja > i) ? NEGF : r0) * g0;   // mask first, then gate (matches ref)
          c = ((jb > i) ? NEGF : r1) * g1;
        }
        p0v[v] = a; p1v[v] = c;
        tmax[v] = fmaxf(a, c);
      }
#pragma unroll
      for (int v = 0; v < 8; ++v) {
        float x = tmax[v];
        for (int off = 1; off < 16; off <<= 1) x = fmaxf(x, __shfl_xor(x, off, 16));
        tmax[v] = x;
      }
#pragma unroll
      for (int v = 0; v < 8; ++v) {
        float mnew = fmaxf(mS[v], tmax[v]);
        float fac  = __expf(mS[v] - mnew);
        mS[v] = mnew;
#pragma unroll
        for (int nt = 0; nt < 4; ++nt) accS[nt][v] *= fac;
        float e0 = __expf(p0v[v] - mnew);
        float e1 = __expf(p1v[v] - mnew);
        p0v[v] = e0; p1v[v] = e1;
        float rs = e0 + e1;
        for (int off = 1; off < 16; off <<= 1) rs += __shfl_xor(rs, off, 16);
        lS[v] = lS[v] * fac + rs;
      }
      // D-layout -> A-layout via per-wave LDS staging (hi/lo bf16)
#pragma unroll
      for (int v = 0; v < 8; ++v) {
        int m = v + 8 * hv;
        float pa = p0v[v], pb2 = p1v[v];
        __bf16 ha = (__bf16)pa, hb = (__bf16)pb2;
        sPh[wid][m][lm]      = ha;
        sPl[wid][m][lm]      = (__bf16)(pa - (float)ha);
        sPh[wid][m][16 + lm] = hb;
        sPl[wid][m][16 + lm] = (__bf16)(pb2 - (float)hb);
      }
      v16bf ph = ld_frag2(&sPh[wid][lm][hv * 8], &sPh[wid][lm][16 + hv * 8]);
      v16bf pl = ld_frag2(&sPl[wid][lm][hv * 8], &sPl[wid][lm][16 + hv * 8]);
#pragma unroll
      for (int nt = 0; nt < 4; ++nt) {
        accS[nt] = WMMA_BF16(ph, fvh[nt], accS[nt]);
        accS[nt] = WMMA_BF16(ph, fvl[nt], accS[nt]);
        accS[nt] = WMMA_BF16(pl, fvh[nt], accS[nt]);
      }
    };

    flavor(accL, mL, lL, true);
    flavor(accM, mM, lM, false);
    p ^= 1;
  }

  // Combine the two softmax branches; write hi/lo bf16 into [B,T,C]
  float wl = 1.f / (1.f + __expf(-lwp[0]));
  float wg = 1.f / (1.f + __expf(-gwp[0]));
  float wsum = wl + wg;
  wl /= wsum; wg /= wsum;
#pragma unroll
  for (int nt = 0; nt < 4; ++nt)
#pragma unroll
    for (int v = 0; v < 8; ++v) {
      int i = q0 + wid * 16 + v + 8 * hv;
      int d = nt * 16 + lm;
      float o = wl * accL[nt][v] / lL[v] + wg * accM[nt][v] / lM[v];
      size_t dst = ((size_t)b * Tn + i) * Hc + h * HDd + d;
      __bf16 hi = (__bf16)o;
      oh[dst] = hi;
      ol[dst] = (__bf16)(o - (float)hi);
    }
}

// ---------------------------------------------------------------------------
extern "C" void kernel_launch(void* const* d_in, const int* in_sizes, int n_in,
                              void* d_out, int out_size, void* d_ws, size_t ws_size,
                              hipStream_t stream) {
  (void)in_sizes; (void)n_in; (void)out_size; (void)ws_size;
  const float* x      = (const float*)d_in[0];
  const float* w_attn = (const float*)d_in[1];
  const float* b_attn = (const float*)d_in[2];
  const float* w_proj = (const float*)d_in[3];
  const float* b_proj = (const float*)d_in[4];
  const float* w_sel1 = (const float*)d_in[5];
  const float* b_sel1 = (const float*)d_in[6];
  const float* w_sel2 = (const float*)d_in[7];
  const float* b_sel2 = (const float*)d_in[8];
  const float* lw     = (const float*)d_in[9];
  const float* gw     = (const float*)d_in[10];

  char* wsp = (char*)d_ws;
  size_t off = 0;
  auto alloc = [&](size_t bytes) -> char* {
    char* p = wsp + off;
    off = (off + bytes + 255) & ~(size_t)255;
    return p;
  };

  const size_t HEAD_ELEMS = (size_t)Bn * NHn * Tn * HDd;  // 3,145,728

  __bf16* xh   = (__bf16*)alloc((size_t)NTOK * Hc * 2);
  __bf16* xl   = (__bf16*)alloc((size_t)NTOK * Hc * 2);
  __bf16* wah  = (__bf16*)alloc((size_t)Hc * H3 * 2);     // transposed [N][K]
  __bf16* wal  = (__bf16*)alloc((size_t)Hc * H3 * 2);
  __bf16* wph  = (__bf16*)alloc((size_t)Hc * Hc * 2);     // transposed
  __bf16* wpl  = (__bf16*)alloc((size_t)Hc * Hc * 2);
  __bf16* w1h  = (__bf16*)alloc((size_t)Hc * (Hc / 4) * 2); // transposed
  __bf16* w1l  = (__bf16*)alloc((size_t)Hc * (Hc / 4) * 2);
  char*   big  = alloc((size_t)NTOK * H3 * 4);            // fp32 qkv, then reused
  __bf16* qhp  = (__bf16*)alloc(HEAD_ELEMS * 2);
  __bf16* qlp  = (__bf16*)alloc(HEAD_ELEMS * 2);
  __bf16* khp  = (__bf16*)alloc(HEAD_ELEMS * 2);
  __bf16* klp  = (__bf16*)alloc(HEAD_ELEMS * 2);
  __bf16* vthp = (__bf16*)alloc(HEAD_ELEMS * 2);          // transposed per head
  __bf16* vtlp = (__bf16*)alloc(HEAD_ELEMS * 2);
  float*  self = (float*)alloc((size_t)NTOK * 4);

  float*  qkvf  = (float*)big;                            // [NTOK, 3H] fp32
  float*  sel1f = (float*)big;                            // [NTOK, 192] fp32 (after repack)
  __bf16* oah   = (__bf16*)big;                           // attn out hi (after sel2)
  __bf16* oal   = (__bf16*)(big + (size_t)NTOK * Hc * 2); // attn out lo

  // 1) precision splits (weights transposed for async row-major staging)
  k_split<<<(NTOK * Hc + 255) / 256, 256, 0, stream>>>(x, xh, xl, NTOK * Hc);
  k_split_t<<<(Hc * H3 + 255) / 256, 256, 0, stream>>>(w_attn, wah, wal, Hc, H3);
  k_split_t<<<(Hc * Hc + 255) / 256, 256, 0, stream>>>(w_proj, wph, wpl, Hc, Hc);
  k_split_t<<<(Hc * (Hc / 4) + 255) / 256, 256, 0, stream>>>(w_sel1, w1h, w1l, Hc, Hc / 4);

  // 2) qkv projection
  {
    dim3 g(H3 / 64, NTOK / 128);
    k_gemm<<<g, 256, 0, stream>>>(xh, xl, wah, wal, b_attn, qkvf, NTOK, H3, Hc);
  }
  // 3) repack to per-head hi/lo (V transposed)
  k_repack_qkv<<<(int)(((size_t)NTOK * H3 + 255) / 256), 256, 0, stream>>>(
      qkvf, qhp, qlp, khp, klp, vthp, vtlp);

  // 4) selectivity gate
  {
    dim3 g((Hc / 4) / 64, NTOK / 128);
    k_gemm<<<g, 256, 0, stream>>>(xh, xl, w1h, w1l, b_sel1, sel1f, NTOK, Hc / 4, Hc);
  }
  k_sel2<<<(NTOK + 255) / 256, 256, 0, stream>>>(sel1f, w_sel2, b_sel2, self, NTOK);

  // 5) fused two-softmax attention
  {
    dim3 g(Tn / 64, Bn * NHn);
    k_attn<<<g, 128, 0, stream>>>(qhp, qlp, khp, klp, vthp, vtlp, self, lw, gw, oah, oal);
  }

  // 6) output projection -> d_out
  {
    dim3 g(Hc / 64, NTOK / 128);
    k_gemm<<<g, 256, 0, stream>>>(oah, oal, wph, wpl, b_proj, (float*)d_out, NTOK, Hc, Hc);
  }
}
